// SimpleMPNN_16939351015862
// MI455X (gfx1250) — matrix operand order repaired and checked by reading, not compiled, not run on hardware
//
#include <hip/hip_runtime.h>
#include <hip/hip_bf16.h>
#include <stdint.h>

#define NN 50000
#define NE 400000
#define H  128
#define NL 3
#define NG 64
#define ET 8   // edge tiles (of 16 rows) per block: 25000/8 = 3125 blocks
#define NT 5   // node tiles per block: 3125/5 = 625 blocks

typedef __attribute__((ext_vector_type(16))) _Float16 v16h;
typedef __attribute__((ext_vector_type(8)))  _Float16 v8h;
typedef __attribute__((ext_vector_type(8)))  float    v8f;

union AFrag { v16h v; v8h h8[2]; };

#if defined(__has_builtin)
# if __has_builtin(__builtin_amdgcn_global_load_async_to_lds_b128) && \
     __has_builtin(__builtin_amdgcn_s_wait_asynccnt)
#  define USE_ASYNC 1
# endif
#endif
#ifndef USE_ASYNC
# define USE_ASYNC 0
#endif

#if USE_ASYNC
typedef __attribute__((__vector_size__(4 * sizeof(int)))) int v4i32;
#define ASYNC_CP16(gsrc, ldst) \
    __builtin_amdgcn_global_load_async_to_lds_b128((v4i32*)(void*)(gsrc), (v4i32*)(void*)(ldst), 0, 0)
#define ASYNC_WAIT() __builtin_amdgcn_s_wait_asynccnt(0)
#else
#define ASYNC_CP16(gsrc, ldst) (*(v8h*)(ldst) = *(const v8h*)(gsrc))
#define ASYNC_WAIT() ((void)0)
#endif

// ---------------- weight transpose + f16 cast: src [L][K][128] -> dst [L][128][Kpad]
__global__ void wt_kernel(const float* __restrict__ src, _Float16* __restrict__ dst,
                          int K, int Kpad, long total) {
    long idx = (long)blockIdx.x * 256 + threadIdx.x;
    if (idx >= total) return;
    int k = (int)(idx % Kpad);
    long t = idx / Kpad;
    int n = (int)(t % H);
    int l = (int)(t / H);
    dst[idx] = (k < K) ? (_Float16)src[((long)l * K + k) * H + n] : (_Float16)0.f;
}

__global__ void zero_kernel(float* p, long n) {
    long i = (long)blockIdx.x * 256 + threadIdx.x;
    if (i < n) p[i] = 0.f;
}

__global__ void embed_kernel(const int* __restrict__ z, const float* __restrict__ embed,
                             float* __restrict__ h, _Float16* __restrict__ hf) {
    long idx = (long)blockIdx.x * 256 + threadIdx.x;
    if (idx >= (long)NN * H) return;
    int n = (int)(idx >> 7), c = (int)(idx & (H - 1));
    float v = embed[(long)z[n] * H + c];
    h[idx]  = v;
    hf[idx] = (_Float16)v;
}

// ---------------- edge message kernel: double-buffered staging, weights in VGPRs
__global__ __launch_bounds__(256)
void edge_kernel(const _Float16* __restrict__ hf, const int* __restrict__ ei,
                 const float* __restrict__ ea,
                 const _Float16* __restrict__ w1t, const float* __restrict__ b1,
                 const _Float16* __restrict__ w2t, const float* __restrict__ b2,
                 float* __restrict__ agg) {
    __shared__ __attribute__((aligned(32))) _Float16 Am[2][16 * 272];
    __shared__ __attribute__((aligned(32))) _Float16 Act[16 * 128];
    __shared__ int   s_dst[2][16];
    __shared__ float s_ea[2][16];

    const int tid  = threadIdx.x;
    const int lane = tid & 31;
    const int ncol = (tid >> 5) * 16 + (lane & 15); // output column owned by lane
    const int hilo = lane >> 4;
    const int kgA  = hilo * 16;  // A-frag contiguous per-lane K range (matches B)
    const int kgB  = hilo * 16;  // B-frag contiguous per-lane K range
    const int arow = lane & 15;
    const int srow = tid >> 4;   // staging row (16 threads per row)
    const int c0   = tid & 15;   // staging chunk (8 halfs)

    // hoist B fragments + scalars (invariant over all tiles of this block)
    v16h B1[8], B2[4];
    #pragma unroll
    for (int j = 0; j < 8; ++j) B1[j] = *(const v16h*)&w1t[(long)ncol * 272 + j * 32 + kgB];
    #pragma unroll
    for (int j = 0; j < 4; ++j) B2[j] = *(const v16h*)&w2t[(long)ncol * 128 + j * 32 + kgB];
    const float wea   = (float)w1t[(long)ncol * 272 + 256]; // edge_attr (K=256) column
    const float bias1 = b1[ncol];
    const float bias2 = b2[ncol];

    auto stage = [&](int t) {
        const int buf = t & 1;
        const int e   = (blockIdx.x * ET + t) * 16 + srow;
        const int src = ei[e];
        const int dst = ei[NE + e];
        const _Float16* hs = hf + (long)src * H + c0 * 8;
        const _Float16* hd = hf + (long)dst * H + c0 * 8;
        ASYNC_CP16(hs, &Am[buf][srow * 272 + c0 * 8]);
        ASYNC_CP16(hd, &Am[buf][srow * 272 + 128 + c0 * 8]);
        if (c0 == 0) { s_dst[buf][srow] = dst; s_ea[buf][srow] = ea[e]; }
        if (c0 == 2 && t + 1 < ET) {
            __builtin_prefetch(&ei[e + 16], 0, 0);
            __builtin_prefetch(&ei[NE + e + 16], 0, 0);
        }
    };

    stage(0);
    for (int t = 0; t < ET; ++t) {
        const int buf = t & 1;
        ASYNC_WAIT();
        __syncthreads();                     // Am[buf] / s_*[buf] ready everywhere

        v8f acc = {};
        #pragma unroll
        for (int r = 0; r < 8; ++r) acc[r] = bias1 + s_ea[buf][r + 8 * hilo] * wea;

        #pragma unroll
        for (int g = 0; g < 8; g += 4) {     // load 4 A-frags ahead, then 4 WMMAs
            AFrag a[4];
            #pragma unroll
            for (int j = 0; j < 4; ++j) {
                a[j].h8[0] = *(const v8h*)&Am[buf][arow * 272 + (g + j) * 32 + kgA];
                a[j].h8[1] = *(const v8h*)&Am[buf][arow * 272 + (g + j) * 32 + 8 + kgA];
            }
            #pragma unroll
            for (int j = 0; j < 4; ++j)
                acc = __builtin_amdgcn_wmma_f32_16x16x32_f16(false, a[j].v, false, B1[g + j], (short)0, acc, false, false);
        }

        if (t + 1 < ET) stage(t + 1);        // overlap next gather with rest of tile

        #pragma unroll
        for (int r = 0; r < 8; ++r) {
            float v = acc[r] > 0.f ? acc[r] : 0.f;
            Act[(r + 8 * hilo) * 128 + ncol] = (_Float16)v;
        }
        __syncthreads();                     // Act visible

        v8f acc2 = {};
        #pragma unroll
        for (int r = 0; r < 8; ++r) acc2[r] = bias2;
        {
            AFrag a[4];
            #pragma unroll
            for (int j = 0; j < 4; ++j) {
                a[j].h8[0] = *(const v8h*)&Act[arow * 128 + j * 32 + kgA];
                a[j].h8[1] = *(const v8h*)&Act[arow * 128 + j * 32 + 8 + kgA];
            }
            #pragma unroll
            for (int j = 0; j < 4; ++j)
                acc2 = __builtin_amdgcn_wmma_f32_16x16x32_f16(false, a[j].v, false, B2[j], (short)0, acc2, false, false);
        }
        #pragma unroll
        for (int r = 0; r < 8; ++r)
            atomicAdd(&agg[(long)s_dst[buf][r + 8 * hilo] * H + ncol], acc2[r]);
    }
}

// ---------------- node update kernel: h += MLP([h | agg]); same pipelined structure
__global__ __launch_bounds__(256)
void node_kernel(float* __restrict__ h, _Float16* __restrict__ hf,
                 const float* __restrict__ agg,
                 const _Float16* __restrict__ w1t, const float* __restrict__ b1,
                 const _Float16* __restrict__ w2t, const float* __restrict__ b2) {
    __shared__ __attribute__((aligned(32))) _Float16 Am[2][16 * 256];
    __shared__ __attribute__((aligned(32))) _Float16 Act[16 * 128];

    const int tid  = threadIdx.x;
    const int lane = tid & 31;
    const int ncol = (tid >> 5) * 16 + (lane & 15);
    const int hilo = lane >> 4;
    const int kgA  = hilo * 16;
    const int kgB  = hilo * 16;
    const int arow = lane & 15;
    const int srow = tid >> 4;
    const int c0   = tid & 15;

    v16h B1[8], B2[4];
    #pragma unroll
    for (int j = 0; j < 8; ++j) B1[j] = *(const v16h*)&w1t[(long)ncol * 256 + j * 32 + kgB];
    #pragma unroll
    for (int j = 0; j < 4; ++j) B2[j] = *(const v16h*)&w2t[(long)ncol * 128 + j * 32 + kgB];
    const float bias1 = b1[ncol];
    const float bias2 = b2[ncol];

    auto stage = [&](int t) {
        const int buf = t & 1;
        const int n0  = (blockIdx.x * NT + t) * 16;
        const _Float16* hp = hf + (long)(n0 + srow) * H + c0 * 8;
        ASYNC_CP16(hp, &Am[buf][srow * 256 + c0 * 8]);
        const float* ap = agg + (long)(n0 + srow) * H;
        #pragma unroll
        for (int i = 0; i < 8; ++i) {
            int c = c0 + i * 16;
            Am[buf][srow * 256 + 128 + c] = (_Float16)ap[c];
        }
    };

    stage(0);
    for (int t = 0; t < NT; ++t) {
        const int buf = t & 1;
        const int n0  = (blockIdx.x * NT + t) * 16;
        ASYNC_WAIT();
        __syncthreads();

        v8f acc = {};
        #pragma unroll
        for (int r = 0; r < 8; ++r) acc[r] = bias1;
        #pragma unroll
        for (int g = 0; g < 8; g += 4) {
            AFrag a[4];
            #pragma unroll
            for (int j = 0; j < 4; ++j) {
                a[j].h8[0] = *(const v8h*)&Am[buf][arow * 256 + (g + j) * 32 + kgA];
                a[j].h8[1] = *(const v8h*)&Am[buf][arow * 256 + (g + j) * 32 + 8 + kgA];
            }
            #pragma unroll
            for (int j = 0; j < 4; ++j)
                acc = __builtin_amdgcn_wmma_f32_16x16x32_f16(false, a[j].v, false, B1[g + j], (short)0, acc, false, false);
        }

        if (t + 1 < NT) stage(t + 1);

        #pragma unroll
        for (int r = 0; r < 8; ++r) {
            float v = acc[r] > 0.f ? acc[r] : 0.f;
            Act[(r + 8 * hilo) * 128 + ncol] = (_Float16)v;
        }
        __syncthreads();

        v8f acc2 = {};
        #pragma unroll
        for (int r = 0; r < 8; ++r) acc2[r] = bias2;
        {
            AFrag a[4];
            #pragma unroll
            for (int j = 0; j < 4; ++j) {
                a[j].h8[0] = *(const v8h*)&Act[arow * 128 + j * 32 + kgA];
                a[j].h8[1] = *(const v8h*)&Act[arow * 128 + j * 32 + 8 + kgA];
            }
            #pragma unroll
            for (int j = 0; j < 4; ++j)
                acc2 = __builtin_amdgcn_wmma_f32_16x16x32_f16(false, a[j].v, false, B2[j], (short)0, acc2, false, false);
        }
        #pragma unroll
        for (int r = 0; r < 8; ++r) {
            long idx = (long)(n0 + r + 8 * hilo) * H + ncol;
            float v = h[idx] + acc2[r];  // residual in f32 master
            h[idx]  = v;
            hf[idx] = (_Float16)v;
        }
    }
}

// ---------------- mean pooling via atomics
__global__ void pool_kernel(const float* __restrict__ h, const int* __restrict__ batch,
                            float* __restrict__ pooled, float* __restrict__ counts) {
    long idx = (long)blockIdx.x * 256 + threadIdx.x;
    if (idx >= (long)NN * H) return;
    int n = (int)(idx >> 7), c = (int)(idx & (H - 1));
    int b = batch[n];
    atomicAdd(&pooled[(long)b * H + c], h[idx]);
    if (c == 0) atomicAdd(&counts[b], 1.0f);
}

// ---------------- tiny readout head: [64,128] -> relu -> [64,1]
__global__ __launch_bounds__(128)
void head_kernel(const float* __restrict__ pooled, const float* __restrict__ counts,
                 const float* __restrict__ w1, const float* __restrict__ b1,
                 const float* __restrict__ w2, const float* __restrict__ b2,
                 float* __restrict__ out) {
    __shared__ float sp[128];
    __shared__ float red[128];
    int g = blockIdx.x, j = threadIdx.x;
    float cnt = counts[g];
    cnt = cnt > 1.f ? cnt : 1.f;
    sp[j] = pooled[(long)g * H + j] / cnt;
    __syncthreads();
    float t = b1[j];
    #pragma unroll 4
    for (int i = 0; i < H; ++i) t += sp[i] * w1[(long)i * H + j];
    t = t > 0.f ? t : 0.f;
    red[j] = t * w2[j];
    __syncthreads();
    for (int s = 64; s > 0; s >>= 1) {
        if (j < s) red[j] += red[j + s];
        __syncthreads();
    }
    if (j == 0) out[g] = red[0] + b2[0];
}

extern "C" void kernel_launch(void* const* d_in, const int* in_sizes, int n_in,
                              void* d_out, int out_size, void* d_ws, size_t ws_size,
                              hipStream_t stream) {
    (void)in_sizes; (void)n_in; (void)out_size; (void)ws_size;
    const int*   z        = (const int*)d_in[0];
    const int*   ei       = (const int*)d_in[1];
    const float* ea       = (const float*)d_in[2];
    const int*   batch    = (const int*)d_in[3];
    const float* embed    = (const float*)d_in[4];
    const float* msg_w1   = (const float*)d_in[5];
    const float* msg_b1   = (const float*)d_in[6];
    const float* msg_w2   = (const float*)d_in[7];
    const float* msg_b2   = (const float*)d_in[8];
    const float* upd_w1   = (const float*)d_in[9];
    const float* upd_b1   = (const float*)d_in[10];
    const float* upd_w2   = (const float*)d_in[11];
    const float* upd_b2   = (const float*)d_in[12];
    const float* head_w1  = (const float*)d_in[13];
    const float* head_b1  = (const float*)d_in[14];
    const float* head_w2  = (const float*)d_in[15];
    const float* head_b2  = (const float*)d_in[16];
    float* out = (float*)d_out;

    // workspace carve-up
    char* ws = (char*)d_ws;
    size_t off = 0;
    auto carve = [&](size_t bytes) { size_t o = off; off = (off + bytes + 511) & ~(size_t)511; return o; };
    float*     h_buf  = (float*)(ws + carve((size_t)NN * H * 4));
    _Float16*  hf_buf = (_Float16*)(ws + carve((size_t)NN * H * 2));
    float*     agg    = (float*)(ws + carve((size_t)NN * H * 4));
    float*     pooled = (float*)(ws + carve((size_t)NG * H * 4));
    float*     counts = (float*)(ws + carve((size_t)NG * 4));
    _Float16*  m1t    = (_Float16*)(ws + carve((size_t)NL * H * 272 * 2)); // msg_w1^T, Kpad=272
    _Float16*  m2t    = (_Float16*)(ws + carve((size_t)NL * H * 128 * 2)); // msg_w2^T
    _Float16*  u1t    = (_Float16*)(ws + carve((size_t)NL * H * 256 * 2)); // upd_w1^T
    _Float16*  u2t    = (_Float16*)(ws + carve((size_t)NL * H * 128 * 2)); // upd_w2^T

    auto blocks = [](long n) { return (int)((n + 255) / 256); };

    // 1) cast/transpose weights to [L][out][Kpad] f16
    long t1 = (long)NL * H * 272, t2 = (long)NL * H * 128, t3 = (long)NL * H * 256;
    wt_kernel<<<blocks(t1), 256, 0, stream>>>(msg_w1, m1t, 257, 272, t1);
    wt_kernel<<<blocks(t2), 256, 0, stream>>>(msg_w2, m2t, 128, 128, t2);
    wt_kernel<<<blocks(t3), 256, 0, stream>>>(upd_w1, u1t, 256, 256, t3);
    wt_kernel<<<blocks(t2), 256, 0, stream>>>(upd_w2, u2t, 128, 128, t2);

    // 2) embedding gather (f32 master + f16 mirror)
    embed_kernel<<<blocks((long)NN * H), 256, 0, stream>>>(z, embed, h_buf, hf_buf);

    // 3) message-passing layers
    for (int l = 0; l < NL; ++l) {
        zero_kernel<<<blocks((long)NN * H), 256, 0, stream>>>(agg, (long)NN * H);
        edge_kernel<<<NE / (16 * ET), 256, 0, stream>>>(
            hf_buf, ei, ea,
            m1t + (size_t)l * H * 272, msg_b1 + (size_t)l * H,
            m2t + (size_t)l * H * 128, msg_b2 + (size_t)l * H,
            agg);
        node_kernel<<<NN / (16 * NT), 256, 0, stream>>>(
            h_buf, hf_buf, agg,
            u1t + (size_t)l * H * 256, upd_b1 + (size_t)l * H,
            u2t + (size_t)l * H * 128, upd_b2 + (size_t)l * H);
    }

    // 4) mean pool + head
    zero_kernel<<<blocks((long)NG * H), 256, 0, stream>>>(pooled, (long)NG * H);
    zero_kernel<<<1, 64, 0, stream>>>(counts, NG);
    pool_kernel<<<blocks((long)NN * H), 256, 0, stream>>>(h_buf, batch, pooled, counts);
    head_kernel<<<NG, 128, 0, stream>>>(pooled, counts, head_w1, head_b1, head_w2, head_b2, out);
}
